// MaskedStabilizedCrossAttention_4105988735325
// MI455X (gfx1250) — matrix operand order repaired and checked by reading, not compiled, not run on hardware
//
#include <hip/hip_runtime.h>
#include <hip/hip_bf16.h>

// ---------- CDNA5 WMMA types ----------
typedef __attribute__((ext_vector_type(16))) __bf16 v16bf;
typedef __attribute__((ext_vector_type(8)))  float  v8f;

union FragB { v16bf v; float4 f4[2]; };
union FragC { v8f v; float f[8]; };

__device__ __forceinline__ unsigned short f2bf(float x) {
    unsigned int u = __float_as_uint(x);
    u += 0x7FFFu + ((u >> 16) & 1u);   // round-to-nearest-even
    return (unsigned short)(u >> 16);
}

// ---------- pack: f32 -> bf16 ----------
__global__ void k_f32_to_bf16(const float* __restrict__ src,
                              unsigned short* __restrict__ dst, int n) {
    int i = (blockIdx.x * blockDim.x + threadIdx.x) * 4;
    if (i + 4 <= n) {
        float4 v = *(const float4*)(src + i);
        unsigned int p0 = (unsigned int)f2bf(v.x) | ((unsigned int)f2bf(v.y) << 16);
        unsigned int p1 = (unsigned int)f2bf(v.z) | ((unsigned int)f2bf(v.w) << 16);
        *(uint2*)(dst + i) = make_uint2(p0, p1);
    }
}

// ---------- pack: W (f32 [1024][1024]) -> W^T bf16 (WT[n][k] = W[k][n]) ----------
__global__ void k_transpose_w(const float* __restrict__ W,
                              unsigned short* __restrict__ WT) {
    int n = blockIdx.x * 16 + threadIdx.x;
    int k = blockIdx.y * 16 + threadIdx.y;
    WT[(size_t)n * 1024 + k] = f2bf(W[(size_t)k * 1024 + n]);
}

// ---------- generic bf16 WMMA GEMM: C[M x 1024] = A[M x 1024] * W + bias ----------
// mode 0: bf16 row-major out; mode 1: bf16 per-head-transposed V out [B,NH,64,512];
// mode 2: f32 row-major out (final projection into d_out)
__global__ __launch_bounds__(256)
void k_gemm(const unsigned short* __restrict__ A,
            const unsigned short* __restrict__ WT,
            const float* __restrict__ bias,
            void* __restrict__ outp, int mode) {
    const int lane = threadIdx.x & 31;
    const int wave = threadIdx.x >> 5;
    const int lr = lane & 15;
    const int hi = lane >> 4;
    const int m0 = blockIdx.y * 64 + (wave & 3) * 16;
    const int nb = blockIdx.x * 64 + (wave >> 2) * 32;

    FragC c0, c1;
#pragma unroll
    for (int i = 0; i < 8; ++i) { c0.f[i] = 0.f; c1.f[i] = 0.f; }

    const unsigned short* arow  = A  + (size_t)(m0 + lr) * 1024;
    const unsigned short* b0row = WT + (size_t)(nb + lr) * 1024;
    const unsigned short* b1row = WT + (size_t)(nb + 16 + lr) * 1024;

    for (int kk = 0; kk < 1024; kk += 32) {
        FragB a, b0, b1;
        const int ak = kk + hi * 16;   // A: lanes 0-15 K 0..15, lanes 16-31 K 16..31
        a.f4[0] = *(const float4*)(arow + ak);
        a.f4[1] = *(const float4*)(arow + ak + 8);
        const int bk = kk + hi * 16;   // B^T row = output column, contiguous 16 K-values
        b0.f4[0] = *(const float4*)(b0row + bk);
        b0.f4[1] = *(const float4*)(b0row + bk + 8);
        b1.f4[0] = *(const float4*)(b1row + bk);
        b1.f4[1] = *(const float4*)(b1row + bk + 8);
        c0.v = __builtin_amdgcn_wmma_f32_16x16x32_bf16(false, a.v, false, b0.v,
                                                       (short)0, c0.v, false, false);
        c1.v = __builtin_amdgcn_wmma_f32_16x16x32_bf16(false, a.v, false, b1.v,
                                                       (short)0, c1.v, false, false);
    }

#pragma unroll
    for (int t = 0; t < 2; ++t) {
        FragC& c = t ? c1 : c0;
        const int col = nb + t * 16 + lr;
        const float bcol = bias[col];
#pragma unroll
        for (int r = 0; r < 8; ++r) {
            const int row = m0 + hi * 8 + r;    // C layout: lanes 0-15 -> M=r, 16-31 -> M=8+r
            const float v = c.f[r] + bcol;
            if (mode == 0) {
                ((unsigned short*)outp)[(size_t)row * 1024 + col] = f2bf(v);
            } else if (mode == 1) {
                const int b = row >> 9, s = row & 511;
                const int h = col >> 6, d = col & 63;
                ((unsigned short*)outp)[(((size_t)(b * 16 + h) * 64 + d) << 9) + s] = f2bf(v);
            } else {
                ((float*)outp)[(size_t)row * 1024 + col] = v;
            }
        }
    }
}

// ---------- fused attention: scores -> bias/mask -> softmax -> attn out + attn*V ----------
__global__ __launch_bounds__(256)
void k_attn(const unsigned short* __restrict__ qbuf,   // [B*512, 1024] bf16
            const unsigned short* __restrict__ kbuf,   // [B*512, 1024] bf16
            const unsigned short* __restrict__ vt,     // [B,NH,64,512] bf16
            unsigned short* __restrict__ att_out,      // [B*512, 1024] bf16
            float* __restrict__ attn_out,              // [B,NH,512,512] f32
            const int* __restrict__ kpm,
            const float* __restrict__ rel_table,
            const float* __restrict__ local_bias,
            const float* __restrict__ log_scale,
            const float* __restrict__ attention_bias) {
    __shared__ unsigned short attn_lds[16 * 512];
    __shared__ float red[8 * 16];

    const int lane = threadIdx.x & 31;
    const int wave = threadIdx.x >> 5;
    const int lr = lane & 15;
    const int hi = lane >> 4;
    const int qt = blockIdx.x, h = blockIdx.y, b = blockIdx.z;
    const int m0 = qt * 16;
    const int nbase = wave * 64;

    const float scale = fminf(fmaxf(__expf(log_scale[0]), 1.0f), 4.0f);
    const float abias = attention_bias[0];

    // ---- scores: Q(16x64) * K^T(64x512), this wave does 16x64 of it ----
    const unsigned short* qrow = qbuf + (size_t)(b * 512 + m0 + lr) * 1024 + h * 64;
    FragC acc[4];
#pragma unroll
    for (int t = 0; t < 4; ++t)
#pragma unroll
        for (int i = 0; i < 8; ++i) acc[t].f[i] = 0.f;

#pragma unroll
    for (int k0 = 0; k0 < 64; k0 += 32) {
        FragB a;
        a.f4[0] = *(const float4*)(qrow + k0 + hi * 16);
        a.f4[1] = *(const float4*)(qrow + k0 + hi * 16 + 8);
#pragma unroll
        for (int tt = 0; tt < 4; ++tt) {
            const int c = nbase + tt * 16 + lr;
            const unsigned short* krow = kbuf + (size_t)(b * 512 + c) * 1024 + h * 64;
            FragB bf;
            bf.f4[0] = *(const float4*)(krow + k0 + hi * 16);
            bf.f4[1] = *(const float4*)(krow + k0 + hi * 16 + 8);
            acc[tt].v = __builtin_amdgcn_wmma_f32_16x16x32_bf16(false, a.v, false, bf.v,
                                                                (short)0, acc[tt].v, false, false);
        }
    }

    // ---- biases, scale, mask, clamp ----
    float vals[4][8];
#pragma unroll
    for (int tt = 0; tt < 4; ++tt) {
        const int col = nbase + tt * 16 + lr;
        const int masked = (kpm[b * 512 + col] == 0);
#pragma unroll
        for (int r = 0; r < 8; ++r) {
            const int row = m0 + hi * 8 + r;
            float s = acc[tt].f[r] * 0.125f;            // 1/sqrt(64)
            int rel = row - col;
            rel = rel < -511 ? -511 : (rel > 511 ? 511 : rel);
            s += rel_table[(rel + 511) * 16 + h];
            s += local_bias[(size_t)row * 512 + col] * 0.05f;
            s = s * scale + abias;
            s = masked ? -50.0f : fminf(fmaxf(s, -50.0f), 50.0f);
            vals[tt][r] = s;
        }
    }

    // ---- row max over 512 cols: shfl within 16-lane col group, LDS across waves ----
    float rmax[8];
#pragma unroll
    for (int r = 0; r < 8; ++r)
        rmax[r] = fmaxf(fmaxf(vals[0][r], vals[1][r]), fmaxf(vals[2][r], vals[3][r]));
#pragma unroll
    for (int m = 1; m < 16; m <<= 1)
#pragma unroll
        for (int r = 0; r < 8; ++r)
            rmax[r] = fmaxf(rmax[r], __shfl_xor(rmax[r], m, 16));
    if (lr == 0)
#pragma unroll
        for (int r = 0; r < 8; ++r) red[wave * 16 + hi * 8 + r] = rmax[r];
    __syncthreads();
#pragma unroll
    for (int r = 0; r < 8; ++r) {
        float mm = -1e30f;
#pragma unroll
        for (int w = 0; w < 8; ++w) mm = fmaxf(mm, red[w * 16 + hi * 8 + r]);
        rmax[r] = mm;
    }
    __syncthreads();

    // ---- exp + row sum ----
    float rsum[8];
#pragma unroll
    for (int r = 0; r < 8; ++r) rsum[r] = 0.f;
#pragma unroll
    for (int tt = 0; tt < 4; ++tt)
#pragma unroll
        for (int r = 0; r < 8; ++r) {
            vals[tt][r] = __expf(vals[tt][r] - rmax[r]);
            rsum[r] += vals[tt][r];
        }
#pragma unroll
    for (int m = 1; m < 16; m <<= 1)
#pragma unroll
        for (int r = 0; r < 8; ++r)
            rsum[r] += __shfl_xor(rsum[r], m, 16);
    if (lr == 0)
#pragma unroll
        for (int r = 0; r < 8; ++r) red[wave * 16 + hi * 8 + r] = rsum[r];
    __syncthreads();
#pragma unroll
    for (int r = 0; r < 8; ++r) {
        float ss = 0.f;
#pragma unroll
        for (int w = 0; w < 8; ++w) ss += red[w * 16 + hi * 8 + r];
        rsum[r] = 1.0f / ss;
    }

    // ---- normalize, write attn (f32) + stash bf16 probs in LDS ----
    float* ap = attn_out + (size_t)(b * 16 + h) * 512 * 512;
#pragma unroll
    for (int tt = 0; tt < 4; ++tt) {
        const int col = nbase + tt * 16 + lr;
#pragma unroll
        for (int r = 0; r < 8; ++r) {
            const int row = m0 + hi * 8 + r;
            const float p = vals[tt][r] * rsum[r];
            ap[(size_t)row * 512 + col] = p;
            attn_lds[(hi * 8 + r) * 512 + col] = f2bf(p);
        }
    }
    __syncthreads();

    // ---- out16 = attn(16x512) * V(512x64); waves 0-3, one 16x16 N-tile each ----
    if (wave < 4) {
        const int d0 = wave * 16;
        FragC o;
#pragma unroll
        for (int i = 0; i < 8; ++i) o.f[i] = 0.f;
        const unsigned short* vrow  = vt + ((size_t)(b * 16 + h) * 64 + d0 + lr) * 512;
        const unsigned short* alrow = attn_lds + lr * 512;
        for (int kk = 0; kk < 512; kk += 32) {
            FragB a, bv;
            a.f4[0]  = *(const float4*)(alrow + kk + hi * 16);
            a.f4[1]  = *(const float4*)(alrow + kk + hi * 16 + 8);
            bv.f4[0] = *(const float4*)(vrow + kk + hi * 16);
            bv.f4[1] = *(const float4*)(vrow + kk + hi * 16 + 8);
            o.v = __builtin_amdgcn_wmma_f32_16x16x32_bf16(false, a.v, false, bv.v,
                                                          (short)0, o.v, false, false);
        }
#pragma unroll
        for (int r = 0; r < 8; ++r) {
            const int row = m0 + hi * 8 + r;
            const int col = h * 64 + d0 + lr;
            att_out[(size_t)(b * 512 + row) * 1024 + col] = f2bf(o.f[r]);
        }
    }
}

// ---------- host-side launch ----------
extern "C" void kernel_launch(void* const* d_in, const int* in_sizes, int n_in,
                              void* d_out, int out_size, void* d_ws, size_t ws_size,
                              hipStream_t stream) {
    (void)in_sizes; (void)n_in; (void)out_size; (void)ws_size;

    const float* query = (const float*)d_in[0];
    const float* key   = (const float*)d_in[1];
    const float* value = (const float*)d_in[2];
    const int*   kpm   = (const int*)d_in[3];
    const float* Wq = (const float*)d_in[4];
    const float* bq = (const float*)d_in[5];
    const float* Wk = (const float*)d_in[6];
    const float* bk = (const float*)d_in[7];
    const float* Wv = (const float*)d_in[8];
    const float* bv = (const float*)d_in[9];
    const float* Wo = (const float*)d_in[10];
    const float* bo = (const float*)d_in[11];
    const float* rel_table      = (const float*)d_in[12];
    const float* local_bias     = (const float*)d_in[13];
    const float* log_scale      = (const float*)d_in[14];
    const float* attention_bias = (const float*)d_in[15];

    char* ws = (char*)d_ws;
    const size_t SZ = (size_t)4096 * 1024 * 2;   // one bf16 [4096,1024] buffer = 8 MB
    unsigned short* qbuf = (unsigned short*)(ws + 0 * SZ);
    unsigned short* kbuf = (unsigned short*)(ws + 1 * SZ);
    unsigned short* vtb  = (unsigned short*)(ws + 2 * SZ);
    unsigned short* xq   = (unsigned short*)(ws + 3 * SZ);
    unsigned short* xk   = (unsigned short*)(ws + 4 * SZ);
    unsigned short* xv   = (unsigned short*)(ws + 5 * SZ);
    unsigned short* att  = (unsigned short*)(ws + 6 * SZ);
    unsigned short* WqT  = (unsigned short*)(ws + 7 * SZ);
    unsigned short* WkT  = WqT + (size_t)1024 * 1024;
    unsigned short* WvT  = WkT + (size_t)1024 * 1024;
    unsigned short* WoT  = WvT + (size_t)1024 * 1024;

    float* out  = (float*)d_out;                       // [8,512,1024]
    float* attn = out + (size_t)8 * 512 * 1024;        // [8,16,512,512]

    const int n = 8 * 512 * 1024;
    k_f32_to_bf16<<<n / 1024, 256, 0, stream>>>(query, xq, n);
    k_f32_to_bf16<<<n / 1024, 256, 0, stream>>>(key,   xk, n);
    k_f32_to_bf16<<<n / 1024, 256, 0, stream>>>(value, xv, n);

    dim3 tg(64, 64), tb(16, 16);
    k_transpose_w<<<tg, tb, 0, stream>>>(Wq, WqT);
    k_transpose_w<<<tg, tb, 0, stream>>>(Wk, WkT);
    k_transpose_w<<<tg, tb, 0, stream>>>(Wv, WvT);
    k_transpose_w<<<tg, tb, 0, stream>>>(Wo, WoT);

    dim3 gg(16, 64);
    k_gemm<<<gg, 256, 0, stream>>>(xq, WqT, bq, qbuf, 0);
    k_gemm<<<gg, 256, 0, stream>>>(xk, WkT, bk, kbuf, 0);
    k_gemm<<<gg, 256, 0, stream>>>(xv, WvT, bv, vtb, 1);

    dim3 ag(32, 16, 8);
    k_attn<<<ag, 256, 0, stream>>>(qbuf, kbuf, vtb, att, attn, kpm,
                                   rel_table, local_bias, log_scale, attention_bias);

    k_gemm<<<gg, 256, 0, stream>>>(att, WoT, bo, out, 2);
}
